// ResonanceAttention_64836826300708
// MI455X (gfx1250) — compile-verified, matrix operand order checked by reading, not run on hardware
//
#include <hip/hip_runtime.h>
#include <hip/hip_bf16.h>
#include <math.h>
#include <stdint.h>

// ---------------- problem constants ----------------
#define BB 2
#define LL 2048
#define DD 1024
#define HH 16
#define DH 64
#define M1 (BB * LL)     // 4096 rows of x / attnout
#define N1 (3 * DD)      // 3072 qkv cols
#define K1 DD            // 1024

typedef __attribute__((ext_vector_type(16))) __bf16 v16bf;
typedef __attribute__((ext_vector_type(8)))  __bf16 v8bf;
typedef __attribute__((ext_vector_type(8)))  float  v8f;

union FragU { v16bf v; v8bf h[2]; };

// Load one 16-bit WMMA operand fragment. Caller passes the per-lane pointer:
//   p = base + (lane&15)*stride + khalf      (khalf = (lane>>4)*8)
// Fragment = 8 contiguous bf16 at p (K=khalf..+7) plus 8 at p+16 (K=khalf+16..+23).
__device__ __forceinline__ v16bf load_frag(const __bf16* p) {
  FragU u;
  u.h[0] = *(const v8bf*)(p);
  u.h[1] = *(const v8bf*)(p + 16);
  return u.v;
}

__device__ __forceinline__ v8f wmma_bf16(v16bf a, v16bf b, v8f c) {
  return __builtin_amdgcn_wmma_f32_16x16x32_bf16(
      /*neg_a=*/false, a, /*neg_b=*/false, b,
      /*c_mod=*/(short)0, c, /*reuse_a=*/false, /*reuse_b=*/false);
}

// ---- CDNA5 async global->LDS copy (ASYNCcnt path) ----
// GVS form: saddr = uniform 64-bit base (SGPR pair), vaddr = per-lane u32 byte
// offset, vdst = per-lane u32 LDS byte address (flat-LDS low 32 bits).
__device__ __forceinline__ void async_copy_b128(uint32_t lds_addr,
                                                const void* gbase,
                                                uint32_t byte_off) {
  asm volatile("global_load_async_to_lds_b128 %0, %1, %2"
               :
               : "v"(lds_addr), "v"(byte_off), "s"(gbase)
               : "memory");
}
#define WAIT_ASYNC(n) asm volatile("s_wait_asynccnt %0" ::"n"(n) : "memory")

// Stage one 128-row x 32-col bf16 slab (global row stride = gstride bytes,
// LDS row stride = 80 bytes, rows padded 32->40 bf16) via 2 async b128 ops
// per thread (4 per wave per A+B slab pair -> asynccnt granularity 4).
__device__ __forceinline__ void issue_slab(const void* gbase, uint32_t gstart,
                                           uint32_t gstride, uint32_t lds_base,
                                           int tid) {
#pragma unroll
  for (int c0 = 0; c0 < 512; c0 += 256) {
    int c = c0 + tid;
    uint32_t row = (uint32_t)(c >> 2);
    uint32_t seg = (uint32_t)(c & 3);
    async_copy_b128(lds_base + row * 80u + seg * 16u,
                    gbase, gstart + row * gstride + seg * 16u);
  }
}

// ---------------- conversion kernels ----------------
__global__ __launch_bounds__(256) void cvt_f32_bf16(const float* __restrict__ src,
                                                    __bf16* __restrict__ dst, int n) {
  int i = blockIdx.x * blockDim.x + threadIdx.x;
  if (i < n) dst[i] = (__bf16)src[i];
}

// WT[n*K + k] = (bf16) W[k*N + n]   (row-major [K,N] -> transposed bf16 [N,K])
__global__ __launch_bounds__(256) void cvt_transpose(const float* __restrict__ W,
                                                     __bf16* __restrict__ WT,
                                                     int K, int N) {
  int i = blockIdx.x * blockDim.x + threadIdx.x;
  if (i >= K * N) return;
  int n = i % N;
  int k = i / N;
  WT[(size_t)n * K + k] = (__bf16)W[i];
}

// ---------------- phase projection ----------------
// phase[b,h,l] = tanh(phase_scale * (x[b,l,:] . W_phase[:,h] + b_phase[h]))
__global__ __launch_bounds__(256) void phase_kernel(const float* __restrict__ x,
                                                    const float* __restrict__ Wp,
                                                    const float* __restrict__ bp,
                                                    const float* __restrict__ p_ps,
                                                    float* __restrict__ phase) {
  int idx = blockIdx.x * blockDim.x + threadIdx.x;   // over B*L*H = 65536
  if (idx >= BB * LL * HH) return;
  int h  = idx & (HH - 1);
  int bl = idx >> 4;
  const float* xr = x + (size_t)bl * DD;
  float acc = bp[h];
  for (int k = 0; k < DD; ++k) acc += xr[k] * Wp[k * HH + h];
  float ps = p_ps[0];
  int b = bl >> 11;            // / LL
  int l = bl & (LL - 1);
  phase[((size_t)(b * HH + h)) * LL + l] = tanhf(ps * acc);
}

// ---------------- qkv projection GEMM (LDS-staged, async, double-buffered) ----
// Block tile 128(M) x 128(N), 8 waves x (16 x 128), K-slabs of 32.
// qkv = x @ W_qkv + b_qkv ; scatter into q[BH,L,DH], k[BH,L,DH], vT[BH,DH,L].
__global__ __launch_bounds__(256) void qkv_gemm(const __bf16* __restrict__ xb,
                                                const __bf16* __restrict__ wT,   // [N1,K1]
                                                const float* __restrict__ bias,
                                                __bf16* __restrict__ qb,
                                                __bf16* __restrict__ kbm,
                                                __bf16* __restrict__ vTb) {
  __shared__ __align__(16) __bf16 As[2][128][40];
  __shared__ __align__(16) __bf16 Bs[2][128][40];
  const int tid = threadIdx.x;
  const int wv = tid >> 5, lane = tid & 31, ln = lane & 15, hi = lane >> 4;
  const int NB = N1 / 128;                       // 24
  const int mb = blockIdx.x / NB, nb = blockIdx.x % NB;

  const uint32_t aRow0 = (uint32_t)(mb * 128) * (uint32_t)(K1 * 2);
  const uint32_t bRow0 = (uint32_t)(nb * 128) * (uint32_t)(K1 * 2);
  const uint32_t ldsA[2] = {(uint32_t)(uintptr_t)&As[0][0][0],
                            (uint32_t)(uintptr_t)&As[1][0][0]};
  const uint32_t ldsB[2] = {(uint32_t)(uintptr_t)&Bs[0][0][0],
                            (uint32_t)(uintptr_t)&Bs[1][0][0]};

  // prologue: slab 0 in flight
  issue_slab(xb, aRow0, (uint32_t)(K1 * 2), ldsA[0], tid);
  issue_slab(wT, bRow0, (uint32_t)(K1 * 2), ldsB[0], tid);

  v8f acc[8];
#pragma unroll
  for (int j = 0; j < 8; ++j) acc[j] = (v8f){};

  const int NK = K1 / 32;                        // 32
  for (int kt = 0; kt < NK; ++kt) {
    const int buf = kt & 1;
    if (kt + 2 < NK) {                           // L2 prefetch of slab kt+2
      int pr = tid & 127;
      const char* pb = (tid < 128) ? (const char*)xb + aRow0
                                   : (const char*)wT + bRow0;
      __builtin_prefetch(pb + (size_t)(kt + 2) * 64 + (size_t)pr * (K1 * 2), 0, 0);
    }
    if (kt + 1 < NK) {                           // stage slab kt+1 into buf^1
      issue_slab(xb, aRow0 + (uint32_t)(kt + 1) * 64u, (uint32_t)(K1 * 2),
                 ldsA[buf ^ 1], tid);
      issue_slab(wT, bRow0 + (uint32_t)(kt + 1) * 64u, (uint32_t)(K1 * 2),
                 ldsB[buf ^ 1], tid);
      WAIT_ASYNC(4);                             // slab kt done; kt+1 in flight
    } else {
      WAIT_ASYNC(0);
    }
    __syncthreads();
    v16bf af = load_frag(&As[buf][wv * 16 + ln][hi * 8]);
#pragma unroll
    for (int j = 0; j < 8; ++j) {
      v16bf bf = load_frag(&Bs[buf][j * 16 + ln][hi * 8]);
      acc[j] = wmma_bf16(af, bf, acc[j]);
    }
    __syncthreads();                             // buf free for reuse
  }

  const int row0 = mb * 128 + wv * 16 + 8 * hi;  // + r
#pragma unroll
  for (int j = 0; j < 8; ++j) {
    int col = nb * 128 + j * 16 + ln;
    float bv = bias[col];
    int s = col >> 10;                           // 0=q 1=k 2=v
    int rem = col & 1023;
    int h = rem >> 6;
    int d = rem & 63;
#pragma unroll
    for (int r = 0; r < 8; ++r) {
      int row = row0 + r;
      int b = row >> 11;
      int l = row & (LL - 1);
      int bh = b * HH + h;
      __bf16 val = (__bf16)(acc[j][r] + bv);
      if (s == 0)      qb [((size_t)bh * LL + l) * DH + d] = val;
      else if (s == 1) kbm[((size_t)bh * LL + l) * DH + d] = val;
      else             vTb[((size_t)bh * DH + d) * LL + l] = val;
    }
  }
}

// ---------------- flash attention with resonance phase penalty ----------------
// One block = 8 waves = 8 consecutive q-tiles of the SAME (b,h). The block
// stages each 32-key K chunk (32xDH) and V^T chunk (DHx32) in LDS once via
// async copies (double-buffered), instead of 8 waves re-reading them.
__global__ __launch_bounds__(256) void attn_kernel(const __bf16* __restrict__ qb,
                                                   const __bf16* __restrict__ kbm,
                                                   const __bf16* __restrict__ vTb,
                                                   const float* __restrict__ phase,
                                                   const int* __restrict__ maskp,
                                                   const float* __restrict__ p_rs,
                                                   __bf16* __restrict__ attnout) {
  __shared__ __align__(16) __bf16 Ks[2][32][72];  // 32 keys x DH(64)+8 pad
  __shared__ __align__(16) __bf16 Vs[2][64][40];  // DH(64) x 32 keys + 8 pad
  __shared__ __align__(16) __bf16 plds[8][16 * 32];
  const int tid  = threadIdx.x;
  const int wib  = tid >> 5;
  const int lane = tid & 31;
  const int ln = lane & 15, hi = lane >> 4;

  const int QT = LL / 16;                 // 128 q-tiles per head
  const int bh = blockIdx.x / (QT / 8);   // uniform per block
  const int qt = (blockIdx.x % (QT / 8)) * 8 + wib;
  const int b  = bh >> 4;                 // / HH
  const int h  = bh & 15;

  const uint32_t ldsK[2] = {(uint32_t)(uintptr_t)&Ks[0][0][0],
                            (uint32_t)(uintptr_t)&Ks[1][0][0]};
  const uint32_t ldsV[2] = {(uint32_t)(uintptr_t)&Vs[0][0][0],
                            (uint32_t)(uintptr_t)&Vs[1][0][0]};
  const uint32_t kStart0 = (uint32_t)((size_t)bh * LL * DH * 2);      // K[bh,0,0]
  const uint32_t vStart0 = (uint32_t)((size_t)bh * DH * LL * 2);      // vT[bh,0,0]
  const uint32_t kr = (uint32_t)(tid >> 3), ksg = (uint32_t)(tid & 7);
  const uint32_t vr = (uint32_t)(tid >> 2), vsg = (uint32_t)(tid & 3);

  // Q A-fragments for K = 0..31 and 32..63 of DH
  size_t qoff = ((size_t)bh * LL + qt * 16 + ln) * DH + hi * 8;
  v16bf qa0 = load_frag(qb + qoff);
  v16bf qa1 = load_frag(qb + qoff + 32);

  const float* ph = phase + (size_t)bh * LL;
  float pq[8], mrow[8], lrow[8];
#pragma unroll
  for (int r = 0; r < 8; ++r) {
    pq[r]   = ph[qt * 16 + r + 8 * hi];
    mrow[r] = -INFINITY;
    lrow[r] = 0.0f;
  }
  v8f o[4];
#pragma unroll
  for (int t = 0; t < 4; ++t) o[t] = (v8f){};

  const float invs = 0.125f;              // 1/sqrt(DH)
  const float rsc  = p_rs[0];
  __bf16* myp = &plds[wib][0];

  // prologue: chunk 0 in flight (1 K + 1 V async b128 per thread -> 2/wave)
  async_copy_b128(ldsK[0] + kr * 144u + ksg * 16u, kbm,
                  kStart0 + kr * (DH * 2) + ksg * 16u);
  async_copy_b128(ldsV[0] + vr * 80u + vsg * 16u, vTb,
                  vStart0 + vr * (LL * 2) + vsg * 16u);

  for (int kc = 0; kc < LL; kc += 32) {
    const int buf = (kc >> 5) & 1;
    if (kc + 64 < LL) {                   // L2 prefetch of chunk kc+64
      __builtin_prefetch((const char*)kbm + kStart0 + (size_t)(kc + 64) * (DH * 2) +
                             (size_t)(tid >> 3) * (DH * 2), 0, 0);
    }
    if (kc + 32 < LL) {                   // stage chunk kc+32 into buf^1
      uint32_t kS = kStart0 + (uint32_t)(kc + 32) * (DH * 2);
      uint32_t vS = vStart0 + (uint32_t)(kc + 32) * 2u;
      async_copy_b128(ldsK[buf ^ 1] + kr * 144u + ksg * 16u, kbm,
                      kS + kr * (DH * 2) + ksg * 16u);
      async_copy_b128(ldsV[buf ^ 1] + vr * 80u + vsg * 16u, vTb,
                      vS + vr * (LL * 2) + vsg * 16u);
      WAIT_ASYNC(2);                      // chunk kc done; kc+32 in flight
    } else {
      WAIT_ASYNC(0);
    }
    __syncthreads();

    // ---- scores: S[16q x 32k] = Q * K^T  (two 16x16 tiles, K-dim = DH = 64)
    v16bf k00 = load_frag(&Ks[buf][ln][hi * 8]);
    v16bf k01 = load_frag(&Ks[buf][ln][hi * 8 + 32]);
    v16bf k10 = load_frag(&Ks[buf][ln + 16][hi * 8]);
    v16bf k11 = load_frag(&Ks[buf][ln + 16][hi * 8 + 32]);
    v8f s0 = (v8f){};
    s0 = wmma_bf16(qa0, k00, s0);
    s0 = wmma_bf16(qa1, k01, s0);
    v8f s1 = (v8f){};
    s1 = wmma_bf16(qa0, k10, s1);
    s1 = wmma_bf16(qa1, k11, s1);

    float pk0 = ph[kc + ln];
    float pk1 = ph[kc + 16 + ln];
    int   mk0 = maskp[b * LL + kc + ln];
    int   mk1 = maskp[b * LL + kc + 16 + ln];

    // ---- resonance penalty + mask + online softmax (rows live in 16-lane halves)
#pragma unroll
    for (int r = 0; r < 8; ++r) {
      float d0 = pq[r] - pk0, d1 = pq[r] - pk1;
      float v0 = s0[r] * invs - rsc * d0 * d0;
      float v1 = s1[r] * invs - rsc * d1 * d1;
      if (mk0 == 0) v0 = -1.0e9f;
      if (mk1 == 0) v1 = -1.0e9f;
      float mx = fmaxf(v0, v1);
      mx = fmaxf(mx, __shfl_xor(mx, 1, 32));
      mx = fmaxf(mx, __shfl_xor(mx, 2, 32));
      mx = fmaxf(mx, __shfl_xor(mx, 4, 32));
      mx = fmaxf(mx, __shfl_xor(mx, 8, 32));
      float mnew  = fmaxf(mrow[r], mx);
      float alpha = __expf(mrow[r] - mnew);
      float p0 = __expf(v0 - mnew);
      float p1 = __expf(v1 - mnew);
      float rs2 = p0 + p1;
      rs2 += __shfl_xor(rs2, 1, 32);
      rs2 += __shfl_xor(rs2, 2, 32);
      rs2 += __shfl_xor(rs2, 4, 32);
      rs2 += __shfl_xor(rs2, 8, 32);
      lrow[r] = lrow[r] * alpha + rs2;
      mrow[r] = mnew;
#pragma unroll
      for (int t = 0; t < 4; ++t) o[t][r] *= alpha;
      // restripe P (C layout: lane=col) -> LDS [row][col] for A-fragment reload
      myp[(r + 8 * hi) * 32 + ln]      = (__bf16)p0;
      myp[(r + 8 * hi) * 32 + 16 + ln] = (__bf16)p1;
    }

    // ---- O += P[16x32] * V[32xDH]  (V^T chunk staged in LDS)
    v16bf pa = load_frag(myp + ln * 32 + hi * 8);
#pragma unroll
    for (int t = 0; t < 4; ++t) {
      v16bf vf = load_frag(&Vs[buf][t * 16 + ln][hi * 8]);
      o[t] = wmma_bf16(pa, vf, o[t]);
    }
    __syncthreads();                      // buf free for reuse
  }

  // ---- normalize and write attnout[B,L,D] (bf16, feeds out-proj A operand)
#pragma unroll
  for (int r = 0; r < 8; ++r) {
    float inv = 1.0f / lrow[r];
    int row = qt * 16 + r + 8 * hi;
    size_t base = ((size_t)b * LL + row) * DD + h * DH;
#pragma unroll
    for (int t = 0; t < 4; ++t)
      attnout[base + t * 16 + ln] = (__bf16)(o[t][r] * inv);
  }
}

// ---------------- output projection GEMM (LDS-staged, async, double-buffered) --
// out = attnout @ W_out + b_out  (f32 result). Block tile 128 x 128.
__global__ __launch_bounds__(256) void out_gemm(const __bf16* __restrict__ ab,
                                                const __bf16* __restrict__ wT,  // [DD,DD]
                                                const float* __restrict__ bias,
                                                float* __restrict__ out) {
  __shared__ __align__(16) __bf16 As[2][128][40];
  __shared__ __align__(16) __bf16 Bs[2][128][40];
  const int tid = threadIdx.x;
  const int wv = tid >> 5, lane = tid & 31, ln = lane & 15, hi = lane >> 4;
  const int NB = DD / 128;                       // 8
  const int mb = blockIdx.x / NB, nb = blockIdx.x % NB;

  const uint32_t aRow0 = (uint32_t)(mb * 128) * (uint32_t)(DD * 2);
  const uint32_t bRow0 = (uint32_t)(nb * 128) * (uint32_t)(DD * 2);
  const uint32_t ldsA[2] = {(uint32_t)(uintptr_t)&As[0][0][0],
                            (uint32_t)(uintptr_t)&As[1][0][0]};
  const uint32_t ldsB[2] = {(uint32_t)(uintptr_t)&Bs[0][0][0],
                            (uint32_t)(uintptr_t)&Bs[1][0][0]};

  issue_slab(ab, aRow0, (uint32_t)(DD * 2), ldsA[0], tid);
  issue_slab(wT, bRow0, (uint32_t)(DD * 2), ldsB[0], tid);

  v8f acc[8];
#pragma unroll
  for (int j = 0; j < 8; ++j) acc[j] = (v8f){};

  const int NK = DD / 32;                        // 32
  for (int kt = 0; kt < NK; ++kt) {
    const int buf = kt & 1;
    if (kt + 2 < NK) {
      int pr = tid & 127;
      const char* pb = (tid < 128) ? (const char*)ab + aRow0
                                   : (const char*)wT + bRow0;
      __builtin_prefetch(pb + (size_t)(kt + 2) * 64 + (size_t)pr * (DD * 2), 0, 0);
    }
    if (kt + 1 < NK) {
      issue_slab(ab, aRow0 + (uint32_t)(kt + 1) * 64u, (uint32_t)(DD * 2),
                 ldsA[buf ^ 1], tid);
      issue_slab(wT, bRow0 + (uint32_t)(kt + 1) * 64u, (uint32_t)(DD * 2),
                 ldsB[buf ^ 1], tid);
      WAIT_ASYNC(4);
    } else {
      WAIT_ASYNC(0);
    }
    __syncthreads();
    v16bf af = load_frag(&As[buf][wv * 16 + ln][hi * 8]);
#pragma unroll
    for (int j = 0; j < 8; ++j) {
      v16bf bf = load_frag(&Bs[buf][j * 16 + ln][hi * 8]);
      acc[j] = wmma_bf16(af, bf, acc[j]);
    }
    __syncthreads();
  }

  const int row0 = mb * 128 + wv * 16 + 8 * hi;
#pragma unroll
  for (int j = 0; j < 8; ++j) {
    int col = nb * 128 + j * 16 + ln;
    float bv = bias[col];
#pragma unroll
    for (int r = 0; r < 8; ++r)
      out[(size_t)(row0 + r) * DD + col] = acc[j][r] + bv;
  }
}

// ---------------- host launcher ----------------
extern "C" void kernel_launch(void* const* d_in, const int* in_sizes, int n_in,
                              void* d_out, int out_size, void* d_ws, size_t ws_size,
                              hipStream_t stream) {
  const float* x      = (const float*)d_in[0];
  const int*   mask   = (const int*)  d_in[1];
  const float* W_qkv  = (const float*)d_in[2];
  const float* b_qkv  = (const float*)d_in[3];
  const float* W_out  = (const float*)d_in[4];
  const float* b_out  = (const float*)d_in[5];
  const float* W_ph   = (const float*)d_in[6];
  const float* b_ph   = (const float*)d_in[7];
  const float* rscale = (const float*)d_in[8];
  const float* pscale = (const float*)d_in[9];
  float* out = (float*)d_out;

  // carve workspace
  char* w = (char*)d_ws;
  size_t off = 0;
  auto take = [&](size_t bytes) -> void* {
    void* p = w + off;
    off += (bytes + 255) & ~(size_t)255;
    return p;
  };
  __bf16* xb      = (__bf16*)take((size_t)M1 * K1 * 2);           // 8 MB
  __bf16* wqkvT   = (__bf16*)take((size_t)N1 * K1 * 2);           // 6 MB
  __bf16* woutT   = (__bf16*)take((size_t)DD * DD * 2);           // 2 MB
  __bf16* qb      = (__bf16*)take((size_t)BB * HH * LL * DH * 2); // 8 MB
  __bf16* kb      = (__bf16*)take((size_t)BB * HH * LL * DH * 2); // 8 MB
  __bf16* vTb     = (__bf16*)take((size_t)BB * HH * DH * LL * 2); // 8 MB
  float*  phase   = (float*) take((size_t)BB * HH * LL * 4);      // 256 KB
  __bf16* attnout = (__bf16*)take((size_t)M1 * DD * 2);           // 8 MB
  (void)ws_size; (void)in_sizes; (void)n_in; (void)out_size;

  const int T = 256;
  // 1) precision conversions / transposes
  cvt_f32_bf16<<<(M1 * K1) / T, T, 0, stream>>>(x, xb, M1 * K1);
  cvt_transpose<<<(K1 * N1) / T, T, 0, stream>>>(W_qkv, wqkvT, K1, N1);
  cvt_transpose<<<(DD * DD) / T, T, 0, stream>>>(W_out, woutT, DD, DD);
  // 2) phase projection (+tanh)
  phase_kernel<<<(BB * LL * HH) / T, T, 0, stream>>>(x, W_ph, b_ph, pscale, phase);
  // 3) qkv projection (WMMA, async LDS pipeline)
  qkv_gemm<<<(M1 / 128) * (N1 / 128), T, 0, stream>>>(xb, wqkvT, b_qkv, qb, kb, vTb);
  // 4) flash attention with resonance penalty (WMMA, async K/V staging)
  attn_kernel<<<(BB * HH * (LL / 16)) / 8, T, 0, stream>>>(qb, kb, vTb, phase, mask,
                                                           rscale, attnout);
  // 5) output projection (WMMA, async LDS pipeline)
  out_gemm<<<(M1 / 128) * (DD / 128), T, 0, stream>>>(attnout, woutT, b_out, out);
}
